// GCN_17308718202892
// MI455X (gfx1250) — matrix operand order repaired and checked by reading, not compiled, not run on hardware
//
#include <hip/hip_runtime.h>
#include <hip/hip_bf16.h>
#include <math.h>

typedef __attribute__((ext_vector_type(16))) __bf16 v16bf;
typedef __attribute__((ext_vector_type(8)))  __bf16 v8bf;
typedef __attribute__((ext_vector_type(4)))  __bf16 v4bf;
typedef __attribute__((ext_vector_type(8)))  float  v8f;

#define DIN 256
#define DH  128
#define DOUT 10
#define DOP  16   // DOUT padded to WMMA tile width

static __device__ __forceinline__ void atomAddF(float* p, float v) {
#if defined(__HIP_PLATFORM_AMD__)
    unsafeAtomicAdd(p, v);   // lowers to global_atomic_add_f32
#else
    atomicAdd(p, v);
#endif
}

// ---------- degree / norm ----------
__global__ void k_fill1(float* deg, int n) {
    int i = blockIdx.x * blockDim.x + threadIdx.x;
    if (i < n) deg[i] = 1.0f;               // self-loop
}
__global__ void k_degcount(const int* __restrict__ ei, float* deg, int e) {
    int i = blockIdx.x * blockDim.x + threadIdx.x;
    if (i < e) atomAddF(&deg[ei[e + i]], 1.0f);   // dst row
}
__global__ void k_rsqrt(float* d, int n) {
    int i = blockIdx.x * blockDim.x + threadIdx.x;
    if (i < n) d[i] = rsqrtf(d[i]);         // deg >= 1 always
}

// ---------- conversions ----------
__global__ void k_cvt_x(const float4* __restrict__ x4, __bf16* __restrict__ xb, int n4) {
    int i = blockIdx.x * blockDim.x + threadIdx.x;
    if (i >= n4) return;
    float4 v = x4[i];
    v4bf o = { (__bf16)v.x, (__bf16)v.y, (__bf16)v.z, (__bf16)v.w };
    *(v4bf*)(xb + 4 * i) = o;
}
__global__ void k_w1t(const float* __restrict__ w1, __bf16* __restrict__ w1t) {
    int i = blockIdx.x * blockDim.x + threadIdx.x;   // i = k*DH + n
    if (i >= DIN * DH) return;
    int k = i / DH, n = i % DH;
    w1t[n * DIN + k] = (__bf16)w1[i];
}
__global__ void k_w2t(const float* __restrict__ w2, __bf16* __restrict__ w2t) {
    int i = blockIdx.x * blockDim.x + threadIdx.x;   // i = n*DH + k (transposed, padded)
    if (i >= DOP * DH) return;
    int n = i / DH, k = i % DH;
    w2t[i] = (n < DOUT) ? (__bf16)w2[k * DOUT + n] : (__bf16)0.0f;
}

// ---------- GEMM1: H[N,128] = Xb[N,256] @ W1 (via W1^T bf16) ----------
__global__ void k_gemm1(const __bf16* __restrict__ xb, const __bf16* __restrict__ w1t,
                        float* __restrict__ H, int nrows) {
    const int lane = threadIdx.x & 31;
    const int wv   = threadIdx.x >> 5;      // 0..7 -> column tile
    const int m0   = blockIdx.x << 4;
    if (m0 >= nrows) return;
    const int n0  = wv << 4;
    const int l15 = lane & 15;
    const int hi  = lane >> 4;
    const __bf16* ap = xb  + (size_t)(m0 + l15) * DIN + hi * 8;   // A: K halves {0-7,16-23}/{8-15,24-31}
    const __bf16* bp = w1t + (size_t)(n0 + l15) * DIN + hi * 8;   // B: K halves {0-7,16-23}/{8-15,24-31}
    v8f acc = {};
#pragma unroll
    for (int k0 = 0; k0 < DIN; k0 += 32) {
        v8bf a0 = *(const v8bf*)(ap + k0);
        v8bf a1 = *(const v8bf*)(ap + k0 + 16);
        v8bf b0 = *(const v8bf*)(bp + k0);
        v8bf b1 = *(const v8bf*)(bp + k0 + 16);
        v16bf A = __builtin_shufflevector(a0, a1, 0,1,2,3,4,5,6,7,8,9,10,11,12,13,14,15);
        v16bf B = __builtin_shufflevector(b0, b1, 0,1,2,3,4,5,6,7,8,9,10,11,12,13,14,15);
        acc = __builtin_amdgcn_wmma_f32_16x16x32_bf16(false, A, false, B, (short)0, acc, false, false);
    }
    float* op = H + (size_t)(m0 + hi * 8) * DH + n0 + l15;
#pragma unroll
    for (int v = 0; v < 8; ++v) op[(size_t)v * DH] = acc[v];
}

// ---------- layer-1 aggregation ----------
__global__ void k_agg1_init(const float4* __restrict__ H4, const float* __restrict__ dinv,
                            float4* __restrict__ agg4, int n32) {
    int i = blockIdx.x * blockDim.x + threadIdx.x;   // i indexes N*32 float4s
    if (i >= n32) return;
    float dd = dinv[i >> 5];
    float s = dd * dd;                               // self-loop norm
    float4 h = H4[i];
    agg4[i] = make_float4(h.x * s, h.y * s, h.z * s, h.w * s);
}
__global__ void k_scatter1(const float* __restrict__ H, const int* __restrict__ ei,
                           const float* __restrict__ dinv, float* __restrict__ agg, int e) {
    int lane = threadIdx.x & 31;
    int edge = blockIdx.x * 8 + (threadIdx.x >> 5);  // one wave per edge
    if (edge >= e) return;
    int s = ei[edge], d = ei[e + edge];
    float nrm = dinv[s] * dinv[d];
    float4 h = *(const float4*)(H + (size_t)s * DH + lane * 4);
    float* ap = agg + (size_t)d * DH + lane * 4;
    atomAddF(ap + 0, h.x * nrm);
    atomAddF(ap + 1, h.y * nrm);
    atomAddF(ap + 2, h.z * nrm);
    atomAddF(ap + 3, h.w * nrm);
}
__global__ void k_relu_cvt(const float4* __restrict__ agg4, const float* __restrict__ b1,
                           __bf16* __restrict__ h1b, int n32) {
    int i = blockIdx.x * blockDim.x + threadIdx.x;
    if (i >= n32) return;
    int j = (i & 31) * 4;
    float4 a = agg4[i];
    v4bf o = { (__bf16)fmaxf(a.x + b1[j + 0], 0.0f),
               (__bf16)fmaxf(a.y + b1[j + 1], 0.0f),
               (__bf16)fmaxf(a.z + b1[j + 2], 0.0f),
               (__bf16)fmaxf(a.w + b1[j + 3], 0.0f) };
    *(v4bf*)(h1b + (size_t)i * 4) = o;
}

// ---------- GEMM2: H2[N,16] = h1b[N,128] @ W2 (via W2^T bf16, padded) ----------
__global__ void k_gemm2(const __bf16* __restrict__ hb, const __bf16* __restrict__ w2t,
                        float* __restrict__ H2, int ntiles) {
    const int lane = threadIdx.x & 31;
    const int tile = blockIdx.x * 8 + (threadIdx.x >> 5);  // 8 row-tiles per block
    if (tile >= ntiles) return;
    const int m0  = tile << 4;
    const int l15 = lane & 15;
    const int hi  = lane >> 4;
    const __bf16* ap = hb  + (size_t)(m0 + l15) * DH + hi * 8;
    const __bf16* bp = w2t + (size_t)l15 * DH + hi * 8;
    v8f acc = {};
#pragma unroll
    for (int k0 = 0; k0 < DH; k0 += 32) {
        v8bf a0 = *(const v8bf*)(ap + k0);
        v8bf a1 = *(const v8bf*)(ap + k0 + 16);
        v8bf b0 = *(const v8bf*)(bp + k0);
        v8bf b1 = *(const v8bf*)(bp + k0 + 16);
        v16bf A = __builtin_shufflevector(a0, a1, 0,1,2,3,4,5,6,7,8,9,10,11,12,13,14,15);
        v16bf B = __builtin_shufflevector(b0, b1, 0,1,2,3,4,5,6,7,8,9,10,11,12,13,14,15);
        acc = __builtin_amdgcn_wmma_f32_16x16x32_bf16(false, A, false, B, (short)0, acc, false, false);
    }
    float* op = H2 + (size_t)(m0 + hi * 8) * DOP + l15;
#pragma unroll
    for (int v = 0; v < 8; ++v) op[(size_t)v * DOP] = acc[v];
}

// ---------- layer-2 aggregation ----------
__global__ void k_agg2_init(const float* __restrict__ H2, const float* __restrict__ dinv,
                            float* __restrict__ out, int n10) {
    int i = blockIdx.x * blockDim.x + threadIdx.x;
    if (i >= n10) return;
    int node = i / DOUT, c = i % DOUT;
    float dd = dinv[node];
    out[i] = H2[(size_t)node * DOP + c] * dd * dd;
}
__global__ void k_scatter2(const float* __restrict__ H2, const int* __restrict__ ei,
                           const float* __restrict__ dinv, float* __restrict__ out, int e) {
    int c = threadIdx.x & 15;
    int edge = blockIdx.x * 16 + (threadIdx.x >> 4);
    if (edge >= e || c >= DOUT) return;
    int s = ei[edge], d = ei[e + edge];
    float nrm = dinv[s] * dinv[d];
    atomAddF(out + (size_t)d * DOUT + c, H2[(size_t)s * DOP + c] * nrm);
}
__global__ void k_bias_logsoftmax(float* __restrict__ out, const float* __restrict__ b2, int n) {
    int i = blockIdx.x * blockDim.x + threadIdx.x;
    if (i >= n) return;
    float v[DOUT];
    float m = -INFINITY;
#pragma unroll
    for (int c = 0; c < DOUT; ++c) { v[c] = out[(size_t)i * DOUT + c] + b2[c]; m = fmaxf(m, v[c]); }
    float s = 0.0f;
#pragma unroll
    for (int c = 0; c < DOUT; ++c) s += __expf(v[c] - m);
    float ls = __logf(s);
    float* o1 = out + (size_t)i * DOUT;
    float* o2 = out + (size_t)n * DOUT + (size_t)i * DOUT;
#pragma unroll
    for (int c = 0; c < DOUT; ++c) { o1[c] = v[c]; o2[c] = v[c] - m - ls; }
}

static inline size_t al256(size_t x) { return (x + 255) & ~(size_t)255; }

extern "C" void kernel_launch(void* const* d_in, const int* in_sizes, int n_in,
                              void* d_out, int out_size, void* d_ws, size_t ws_size,
                              hipStream_t stream) {
    const float* x  = (const float*)d_in[0];
    const int*   ei = (const int*)d_in[1];     // [2,E]: row0=src, row1=dst (JAX int64 -> int32)
    const float* W1 = (const float*)d_in[2];
    const float* b1 = (const float*)d_in[3];
    const float* W2 = (const float*)d_in[4];
    const float* b2 = (const float*)d_in[5];
    float* out = (float*)d_out;

    const int N = in_sizes[0] / DIN;
    const int E = in_sizes[1] / 2;

    // workspace layout (aliased where lifetimes allow)
    char* w = (char*)d_ws;
    size_t o = 0;
    float*  deg  = (float*)(w + o);  o += al256((size_t)N * 4);            // becomes dinv in place
    __bf16* W1T  = (__bf16*)(w + o); o += al256((size_t)DH * DIN * 2);
    __bf16* W2T  = (__bf16*)(w + o); o += al256((size_t)DOP * DH * 2);
    __bf16* xb   = (__bf16*)(w + o); o += al256((size_t)N * DIN * 2);      // 51.2 MB
    float*  H    = (float*)(w + o);  o += al256((size_t)N * DH * 4);       // 51.2 MB
    __bf16* h1b  = (__bf16*)(w + o); o += al256((size_t)N * DH * 2);       // 25.6 MB
    float*  agg1 = (float*)xb;   // alias: xb dead after gemm1
    float*  H2   = H;            // alias: H dead after relu

    float* dinv = deg;
    const int T = 256;

    // norms
    k_fill1   <<<(N + T - 1) / T, T, 0, stream>>>(deg, N);
    k_degcount<<<(E + T - 1) / T, T, 0, stream>>>(ei, deg, E);
    k_rsqrt   <<<(N + T - 1) / T, T, 0, stream>>>(deg, N);

    // bf16 conversions
    int n4 = N * (DIN / 4);
    k_cvt_x<<<(n4 + T - 1) / T, T, 0, stream>>>((const float4*)x, xb, n4);
    k_w1t  <<<(DIN * DH + T - 1) / T, T, 0, stream>>>(W1, W1T);
    k_w2t  <<<(DOP * DH + T - 1) / T, T, 0, stream>>>(W2, W2T);

    // layer 1
    int rowTiles = N / 16;                       // N=100000 -> 6250 exact
    k_gemm1<<<rowTiles, T, 0, stream>>>(xb, W1T, H, N);
    int n32 = N * (DH / 4);
    k_agg1_init<<<(n32 + T - 1) / T, T, 0, stream>>>((const float4*)H, dinv, (float4*)agg1, n32);
    k_scatter1 <<<(E + 7) / 8, T, 0, stream>>>(H, ei, dinv, agg1, E);
    k_relu_cvt <<<(n32 + T - 1) / T, T, 0, stream>>>((const float4*)agg1, b1, h1b, n32);

    // layer 2
    k_gemm2<<<(rowTiles + 7) / 8, T, 0, stream>>>(h1b, W2T, H2, rowTiles);
    int n10 = N * DOUT;
    k_agg2_init<<<(n10 + T - 1) / T, T, 0, stream>>>(H2, dinv, out, n10);
    k_scatter2 <<<(E + 15) / 16, T, 0, stream>>>(H2, ei, dinv, out, E);
    k_bias_logsoftmax<<<(N + T - 1) / T, T, 0, stream>>>(out, b2, N);
}
